// VQVAE_84293028151382
// MI455X (gfx1250) — compile-verified, matrix operand order checked by reading, not run on hardware
//
#include <hip/hip_runtime.h>
#include <hip/hip_bf16.h>
#include <math.h>

// ---------------------------------------------------------------------------
// VQ-VAE forward for MI455X (gfx1250, wave32, WMMA).
// Convs = implicit GEMM with v_wmma_f32_16x16x32_f16 (f16 operands, f32 acc).
// - Weight (B) tiles staged with CDNA5 async global->LDS DMA
//   (global_load_async_to_lds_b128 / ASYNCcnt), overlapped with the VALU
//   im2col staging of the A tile.
// - K panel of 64 -> 8 WMMAs per barrier round (amortizes sync latency,
//   lets XDL WMMAs co-execute with next panel's VALU staging).
// - All index math is shift/mask or 16-bit magic-multiply (no int division).
// - Transposed convs via lhs-dilation in the im2col address generator.
// - VQ / argmin / losses in fp32 VALU for exact semantics.
// ---------------------------------------------------------------------------

typedef __attribute__((ext_vector_type(16))) _Float16 v16h;
typedef __attribute__((ext_vector_type(8)))  _Float16 v8h;
typedef __attribute__((ext_vector_type(8)))  float    v8f;

#define F_RELU_IN  1
#define F_RELU_OUT 2
#define F_BIAS     4
#define F_RES      8

// ---------------------------------------------------------------------------
// Weight prep: fp32 conv weights -> f16 GEMM layout [NP][KP] (N-major rows,
// contiguous K). NP multiple of 64, KP multiple of 64: guard-free B tiles.
// transposed=1: PyTorch ConvTranspose2d layout (Cin, Cout, kh, kw) + flip:
// B[k=(c,r,s)][n] = w[c][n][kh-1-r][kw-1-s].
// ---------------------------------------------------------------------------
__global__ void vqvae_prep_w(const float* __restrict__ w, _Float16* __restrict__ out,
                             int Cout, int Cin, int kh, int kw, int KP, int NP,
                             int transposed)
{
    int idx = blockIdx.x * blockDim.x + threadIdx.x;
    int total = NP * KP;
    if (idx >= total) return;
    int n = idx / KP;
    int k = idx - n * KP;
    int khkw = kh * kw;
    int K = Cin * khkw;
    float v = 0.0f;
    if (n < Cout && k < K) {
        int c = k / khkw;
        int rs = k - c * khkw;
        int r = rs / kw;
        int s = rs - r * kw;
        if (!transposed) {
            v = w[(((size_t)n * Cin + c) * kh + r) * kw + s];
        } else {
            v = w[(((size_t)c * Cout + n) * kh + (kh - 1 - r)) * kw + (kw - 1 - s)];
        }
    }
    out[(size_t)n * KP + k] = (_Float16)v;
}

// ---------------------------------------------------------------------------
// Implicit-GEMM conv. Block = 128 threads = 4 wave32s.
// Block tile: 64 (M = output pixels) x 64 (N = output channels).
// Wave w owns M rows [w*16, w*16+16), 4 WMMA tiles across N.
// K loop in panels of 64 (2 WMMA K-substeps of 32 each).
// woSh/howoSh: log2(Wo), log2(Ho*Wo) (all layers have pow2 spatial dims).
// magicC/magicR: ceil(2^16/(kh*kw)), ceil(2^16/kw) for div-free decode.
// ---------------------------------------------------------------------------
__global__ __launch_bounds__(128)
void vqvae_conv_wmma(const float* __restrict__ in, const _Float16* __restrict__ wgt,
                     const float* __restrict__ bias, const float* __restrict__ res,
                     float* __restrict__ out,
                     int M, int Cin, int Hi, int Wi, int Ho, int Wo,
                     int kh, int kw, int stride, int pad, int dil,
                     int Cout, int KP, int NP, int flags,
                     int woSh, int howoSh, unsigned magicC, unsigned magicR)
{
    __shared__ __align__(16) _Float16 As[64][72];  // [m_local][k_local], 144B rows
    __shared__ __align__(16) _Float16 Bs[64][72];  // [n_local][k_local]

    const int tid  = threadIdx.x;
    const int wave = tid >> 5;
    const int lane = tid & 31;
    const int m0 = blockIdx.x * 64;
    const int n0 = blockIdx.y * 64;

    const int khkw   = kh * kw;
    const int K      = Cin * khkw;
    const int woMask   = (1 << woSh) - 1;
    const int howoMask = (1 << howoSh) - 1;
    const int Hd = dil * (Hi - 1) + 1;   // dilated input extent
    const int Wd = dil * (Wi - 1) + 1;

    v8f acc[4] = {};

    // Per-thread A staging coordinates: row = tid/2, 32-half strip = (tid&1)*32.
    const int arow = tid >> 1;
    const int akl0 = (tid & 1) * 32;
    const int am = m0 + arow;
    const bool amv = (am < M);
    const int ab  = am >> howoSh;
    const int ar2 = am & howoMask;
    const int aoy = ar2 >> woSh;
    const int aox = ar2 & woMask;
    const int ybase = aoy * stride - pad;
    const int xbase = aox * stride - pad;

    // Async B staging: 64x64 halves = 512 16-byte chunks; 4 chunks per thread.
    const unsigned q0 = (unsigned)tid * 4u;
    const unsigned long long wbase = (unsigned long long)(size_t)wgt;

    for (int k0 = 0; k0 < KP; k0 += 64) {
        // ---- kick off async DMA of the B (weight) tile into LDS ----
#pragma unroll
        for (int qq = 0; qq < 4; ++qq) {
            unsigned q   = q0 + (unsigned)qq;
            unsigned row = q >> 3;           // 0..63 (all rows valid: NP % 64 == 0)
            unsigned cc  = (q & 7u) * 8u;    // half index 0,8,...,56
            unsigned ldsDst = (unsigned)(size_t)&Bs[row][cc];
            unsigned goff = (((unsigned)(n0 + (int)row) * (unsigned)KP
                              + (unsigned)k0 + cc) * 2u);
            asm volatile("global_load_async_to_lds_b128 %0, %1, %2"
                         :: "v"(ldsDst), "v"(goff), "s"(wbase)
                         : "memory");
        }

        // ---- stage A tile: im2col + relu_in + fp32->f16 (overlaps DMA) ----
#pragma unroll
        for (int e = 0; e < 32; ++e) {
            int k = k0 + akl0 + e;
            float v = 0.0f;
            if (amv && k < K) {
                int c  = (int)(((unsigned)k * magicC) >> 16);
                int rs = k - c * khkw;
                int r  = (int)(((unsigned)rs * magicR) >> 16);
                int s  = rs - r * kw;
                int yd = ybase + r;
                int xd = xbase + s;
                if (yd >= 0 && xd >= 0 && yd < Hd && xd < Wd) {
                    bool ok = (dil == 1) || (((yd | xd) & 1) == 0);
                    if (ok) {
                        int yi = (dil == 2) ? (yd >> 1) : yd;
                        int xi = (dil == 2) ? (xd >> 1) : xd;
                        v = in[(((size_t)ab * Cin + c) * Hi + yi) * Wi + xi];
                    }
                }
                if (flags & F_RELU_IN) v = v > 0.0f ? v : 0.0f;
            }
            As[arow][akl0 + e] = (_Float16)v;
        }

        // ---- wait for DMA + make both tiles visible workgroup-wide ----
        asm volatile("s_wait_asynccnt 0" ::: "memory");
        __syncthreads();

        // ---- 2 K-substeps x 4 N-tiles = 8 WMMAs per barrier round ----
        const int frow = wave * 16 + (lane & 15);
        const int kb   = (lane >> 4) * 8;  // lanes 0-15: K 0..7/16..23; 16-31: 8..15/24..31
#pragma unroll
        for (int p = 0; p < 2; ++p) {
            const int pb = p * 32;
            v16h afrag;
            {
                v8h lo = *(const v8h*)&As[frow][pb + kb];
                v8h hi = *(const v8h*)&As[frow][pb + 16 + kb];
#pragma unroll
                for (int i = 0; i < 8; ++i) { afrag[i] = lo[i]; afrag[8 + i] = hi[i]; }
            }
#pragma unroll
            for (int t = 0; t < 4; ++t) {
                const int ncol = t * 16 + (lane & 15);
                v16h bfrag;
                v8h lo = *(const v8h*)&Bs[ncol][pb + kb];
                v8h hi = *(const v8h*)&Bs[ncol][pb + 16 + kb];
#pragma unroll
                for (int i = 0; i < 8; ++i) { bfrag[i] = lo[i]; bfrag[8 + i] = hi[i]; }
                acc[t] = __builtin_amdgcn_wmma_f32_16x16x32_f16(
                    false, afrag, false, bfrag, (short)0, acc[t], false, false);
            }
        }
        __syncthreads();
    }

    // ---- epilogue: C/D layout -> NCHW, fuse bias / residual / relu ----
    const int rowbase = m0 + wave * 16 + (lane >> 4) * 8;  // + v gives M row
    const int ncb = (lane & 15);
#pragma unroll
    for (int t = 0; t < 4; ++t) {
        int n = n0 + t * 16 + ncb;
        if (n >= Cout) continue;
        float bv = (flags & F_BIAS) ? bias[n] : 0.0f;
#pragma unroll
        for (int v = 0; v < 8; ++v) {
            int m = rowbase + v;
            if (m >= M) continue;
            int b  = m >> howoSh;
            int r2 = m & howoMask;
            int oy = r2 >> woSh;
            int ox = r2 & woMask;
            size_t oidx = (((size_t)b * Cout + n) * Ho + oy) * Wo + ox;
            float val = acc[t][v] + bv;
            if (flags & F_RES) val += res[oidx];
            if (flags & F_RELU_OUT) val = val > 0.0f ? val : 0.0f;
            out[oidx] = val;
        }
    }
}

// ---------------------------------------------------------------------------
// Vector quantization: one thread per spatial position. fp32 exact distances,
// first-min argmin (matches jnp.argmin). Writes z_q in NCHW, accumulates
// sum((z_q - z)^2) and per-code counts.
// ---------------------------------------------------------------------------
__global__ __launch_bounds__(256)
void vqvae_vq(const float* __restrict__ ze, const float* __restrict__ emb,
              float* __restrict__ zq, float* __restrict__ counts,
              float* __restrict__ acc, int Bn, int C, int HW, int nE)
{
    int i = blockIdx.x * blockDim.x + threadIdx.x;
    int total = Bn * HW;
    if (i >= total) return;
    int b = i / HW;
    int p = i - b * HW;

    float z[64];
    const float* zp = ze + (size_t)b * C * HW + p;
    float zn = 0.0f;
#pragma unroll
    for (int e = 0; e < 64; ++e) {
        float v = zp[(size_t)e * HW];
        z[e] = v;
        zn += v * v;
    }

    int best = 0;
    float bestd = 3.402823466e38f;
    for (int j = 0; j < nE; ++j) {
        const float* ej = emb + (size_t)j * 64;
        float dot = 0.0f, en = 0.0f;
#pragma unroll
        for (int e = 0; e < 64; ++e) {
            float w = ej[e];
            dot += z[e] * w;
            en  += w * w;
        }
        float d = zn + en - 2.0f * dot;
        if (d < bestd) { bestd = d; best = j; }
    }

    const float* eb = emb + (size_t)best * 64;
    float s = 0.0f;
    float* qp = zq + (size_t)b * C * HW + p;
#pragma unroll
    for (int e = 0; e < 64; ++e) {
        float q = eb[e];
        float dlt = q - z[e];
        s += dlt * dlt;
        qp[(size_t)e * HW] = q;
    }
    atomicAdd(&acc[0], s);
    atomicAdd(&counts[best], 1.0f);
}

// ---------------------------------------------------------------------------
// sum((x_hat - x)^2) grid-stride reduction.
// ---------------------------------------------------------------------------
__global__ __launch_bounds__(256)
void vqvae_recon(const float* __restrict__ xh, const float* __restrict__ x,
                 float* __restrict__ acc, int n)
{
    __shared__ float sred[256];
    int i = blockIdx.x * blockDim.x + threadIdx.x;
    float s = 0.0f;
    for (; i < n; i += gridDim.x * blockDim.x) {
        float d = xh[i] - x[i];
        s += d * d;
    }
    sred[threadIdx.x] = s;
    __syncthreads();
    for (int o = 128; o > 0; o >>= 1) {
        if ((int)threadIdx.x < o) sred[threadIdx.x] += sred[threadIdx.x + o];
        __syncthreads();
    }
    if (threadIdx.x == 0) atomicAdd(&acc[1], sred[0]);
}

// ---------------------------------------------------------------------------
// Finalize: perplexity from counts, assemble the 4 scalar outputs.
// ---------------------------------------------------------------------------
__global__ __launch_bounds__(512)
void vqvae_final(const float* __restrict__ counts, const float* __restrict__ acc,
                 float* __restrict__ out, int nE, float invN, float invZ,
                 float invX, float beta)
{
    __shared__ float sred[512];
    int t = threadIdx.x;
    float c = (t < nE) ? counts[t] : 0.0f;
    float em = c * invN;
    sred[t] = em * logf(em + 1e-10f);
    __syncthreads();
    for (int o = 256; o > 0; o >>= 1) {
        if (t < o) sred[t] += sred[t + o];
        __syncthreads();
    }
    if (t == 0) {
        float perp = expf(-sred[0]);
        float embl = (1.0f + beta) * acc[0] * invZ;   // mean((zq-z)^2)*(1+beta)
        float rec  = acc[1] * invX;
        out[0] = rec + embl;  // total_loss
        out[1] = embl;        // emb_loss
        out[2] = rec;         // recon_loss
        out[3] = perp;        // perplexity
    }
}

__global__ void vqvae_zero(float* __restrict__ acc, float* __restrict__ counts, int n)
{
    int i = blockIdx.x * blockDim.x + threadIdx.x;
    if (i < 8) acc[i] = 0.0f;
    if (i < n) counts[i] = 0.0f;
}

// ---------------------------------------------------------------------------
// Host-side orchestration.
// ---------------------------------------------------------------------------
extern "C" void kernel_launch(void* const* d_in, const int* in_sizes, int n_in,
                              void* d_out, int out_size, void* d_ws, size_t ws_size,
                              hipStream_t stream)
{
    (void)in_sizes; (void)n_in; (void)out_size; (void)ws_size;

    const float* x       = (const float*)d_in[0];
    const float* enc_w1  = (const float*)d_in[1];
    const float* enc_b1  = (const float*)d_in[2];
    const float* enc_w2  = (const float*)d_in[3];
    const float* enc_b2  = (const float*)d_in[4];
    const float* enc_w3  = (const float*)d_in[5];
    const float* enc_b3  = (const float*)d_in[6];
    const float* enc_rw1 = (const float*)d_in[7];
    const float* enc_rw2 = (const float*)d_in[8];
    const float* preq_w  = (const float*)d_in[9];
    const float* preq_b  = (const float*)d_in[10];
    const float* emb     = (const float*)d_in[11];
    const float* dec_w1  = (const float*)d_in[12];
    const float* dec_b1  = (const float*)d_in[13];
    const float* dec_rw1 = (const float*)d_in[14];
    const float* dec_rw2 = (const float*)d_in[15];
    const float* dec_w2  = (const float*)d_in[16];
    const float* dec_b2  = (const float*)d_in[17];
    const float* dec_w3  = (const float*)d_in[18];
    const float* dec_b3  = (const float*)d_in[19];

    // Workspace layout (floats). Buffers reused across pipeline phases.
    float* H1  = (float*)d_ws;          // 64x64x64x64   (enc h1, later dec d2)
    float* H2  = H1 + 16777216;         // 64x128x32x32  (enc h2, later dec d1)
    float* H3  = H2 + 8388608;          // 64x128x32x32  (enc h3/res, later x_hat)
    float* RT  = H3 + 8388608;          // 64x32x32x32   (res hidden)
    float* ZE  = RT + 2097152;          // 64x64x32x32
    float* ZQ  = ZE + 4194304;          // 64x64x32x32
    float* ACC = ZQ + 4194304;          // [0]=vq sq sum, [1]=recon sq sum
    float* CNT = ACC + 8;               // 512 counts
    _Float16* WH = (_Float16*)(CNT + 512);

    vqvae_zero<<<2, 256, 0, stream>>>(ACC, CNT, 512);

    struct WDesc { _Float16* p; int KP, NP; };
    _Float16* wp = WH;
    auto prep = [&](const float* w, int Cout, int Cin, int kh, int kw, int transposed) {
        WDesc d;
        d.KP = ((Cin * kh * kw) + 63) & ~63;  // multiple of 64: full K panels
        d.NP = (Cout + 63) & ~63;             // multiple of 64: guard-free B tiles
        d.p  = wp;
        int total = d.NP * d.KP;
        vqvae_prep_w<<<(total + 255) / 256, 256, 0, stream>>>(
            w, d.p, Cout, Cin, kh, kw, d.KP, d.NP, transposed);
        wp += total;
        return d;
    };

    WDesc w_e1 = prep(enc_w1, 64, 3, 4, 4, 0);
    WDesc w_e2 = prep(enc_w2, 128, 64, 4, 4, 0);
    WDesc w_e3 = prep(enc_w3, 128, 128, 3, 3, 0);
    WDesc w_er1[2], w_er2[2], w_dr1[2], w_dr2[2];
    for (int i = 0; i < 2; ++i) {
        w_er1[i] = prep(enc_rw1 + (size_t)i * 32 * 128 * 9, 32, 128, 3, 3, 0);
        w_er2[i] = prep(enc_rw2 + (size_t)i * 128 * 32, 128, 32, 1, 1, 0);
    }
    WDesc w_pq = prep(preq_w, 64, 128, 1, 1, 0);
    WDesc w_d1 = prep(dec_w1, 128, 64, 3, 3, 1);
    for (int i = 0; i < 2; ++i) {
        w_dr1[i] = prep(dec_rw1 + (size_t)i * 32 * 128 * 9, 32, 128, 3, 3, 0);
        w_dr2[i] = prep(dec_rw2 + (size_t)i * 128 * 32, 128, 32, 1, 1, 0);
    }
    WDesc w_d2 = prep(dec_w2, 64, 128, 4, 4, 1);
    WDesc w_d3 = prep(dec_w3, 3, 64, 4, 4, 1);

    auto ilog2 = [](int v) { int s = 0; while ((1 << s) < v) ++s; return s; };
    auto magic16 = [](int d) { return (unsigned)((65536 + d - 1) / d); };

    auto conv = [&](const float* in, WDesc w, const float* bias, const float* res,
                    float* out, int Cin, int Hi, int Wi, int Ho, int Wo,
                    int kh, int kw, int stride, int pad, int dil, int Cout, int flags) {
        int M = 64 * Ho * Wo;
        dim3 grid((unsigned)((M + 63) / 64), (unsigned)(w.NP / 64));
        vqvae_conv_wmma<<<grid, 128, 0, stream>>>(
            in, w.p, bias, res, out, M, Cin, Hi, Wi, Ho, Wo,
            kh, kw, stride, pad, dil, Cout, w.KP, w.NP, flags,
            ilog2(Wo), ilog2(Ho * Wo), magic16(kh * kw), magic16(kw));
    };

    // --- Encoder ---
    conv(x,  w_e1, enc_b1, nullptr, H1, 3, 128, 128, 64, 64, 4, 4, 2, 1, 1, 64,
         F_BIAS | F_RELU_OUT);
    conv(H1, w_e2, enc_b2, nullptr, H2, 64, 64, 64, 32, 32, 4, 4, 2, 1, 1, 128,
         F_BIAS | F_RELU_OUT);
    conv(H2, w_e3, enc_b3, nullptr, H3, 128, 32, 32, 32, 32, 3, 3, 1, 1, 1, 128,
         F_BIAS);
    for (int i = 0; i < 2; ++i) {
        conv(H3, w_er1[i], nullptr, nullptr, RT, 128, 32, 32, 32, 32, 3, 3, 1, 1, 1, 32,
             F_RELU_IN);
        conv(RT, w_er2[i], nullptr, H3, H3, 32, 32, 32, 32, 32, 1, 1, 1, 0, 1, 128,
             F_RELU_IN | F_RES);
    }
    // trailing relu of the res stack fused as RELU_IN of pre-quant conv
    conv(H3, w_pq, preq_b, nullptr, ZE, 128, 32, 32, 32, 32, 1, 1, 1, 0, 1, 64,
         F_BIAS | F_RELU_IN);

    // --- Vector quantization ---
    vqvae_vq<<<(65536 + 255) / 256, 256, 0, stream>>>(ZE, emb, ZQ, CNT, ACC,
                                                      64, 64, 1024, 512);

    // --- Decoder ---
    conv(ZQ, w_d1, dec_b1, nullptr, H2, 64, 32, 32, 32, 32, 3, 3, 1, 1, 1, 128,
         F_BIAS);
    for (int i = 0; i < 2; ++i) {
        conv(H2, w_dr1[i], nullptr, nullptr, RT, 128, 32, 32, 32, 32, 3, 3, 1, 1, 1, 32,
             F_RELU_IN);
        conv(RT, w_dr2[i], nullptr, H2, H2, 32, 32, 32, 32, 32, 1, 1, 1, 0, 1, 128,
             F_RELU_IN | F_RES);
    }
    // ConvTranspose stride 2 -> lhs_dilation 2, pad = k-1-pad = 2, stride 1
    conv(H2, w_d2, dec_b2, nullptr, H1, 128, 32, 32, 64, 64, 4, 4, 1, 2, 2, 64,
         F_BIAS | F_RELU_IN | F_RELU_OUT);
    float* XH = H3;  // h3 dead after pre-quant; reuse for x_hat
    conv(H1, w_d3, dec_b3, nullptr, XH, 64, 64, 64, 128, 128, 4, 4, 1, 2, 2, 3,
         F_BIAS);

    // --- Losses ---
    vqvae_recon<<<1024, 256, 0, stream>>>(XH, x, ACC, 64 * 3 * 128 * 128);
    vqvae_final<<<1, 512, 0, stream>>>(CNT, ACC, (float*)d_out, 512,
                                       1.0f / 65536.0f,
                                       1.0f / (65536.0f * 64.0f),
                                       1.0f / 3145728.0f,
                                       0.25f);
}
